// DualHeadFusion_5703716569569
// MI455X (gfx1250) — matrix operand order repaired and checked by reading, not compiled, not run on hardware
//
#include <hip/hip_runtime.h>
#include <hip/hip_bf16.h>

// ---------------------------------------------------------------------------
// Types for CDNA5 WMMA
// ---------------------------------------------------------------------------
typedef __bf16 v16bf __attribute__((ext_vector_type(16)));
typedef __bf16 v8bf  __attribute__((ext_vector_type(8)));
typedef float  v8f   __attribute__((ext_vector_type(8)));
typedef unsigned short u16;

#define WMMA_BF16(A, B, C) \
  __builtin_amdgcn_wmma_f32_16x16x32_bf16(false, (A), false, (B), (short)0, (C), false, false)

// Native bf16 convert (RNE).
__device__ __forceinline__ u16 f2bf(float f) {
  __bf16 h = (__bf16)f;
  return __builtin_bit_cast(u16, h);
}
// 8-wide f32 -> bf16 convert as a single vector fptrunc: lowers to packed
// v_cvt_pk_bf16_f32 pairs with the output dwords already in order (no
// half-register splicing fixups).
__device__ __forceinline__ uint4 cvt8(const float4 a, const float4 b) {
  v8f f;
  f[0] = a.x; f[1] = a.y; f[2] = a.z; f[3] = a.w;
  f[4] = b.x; f[5] = b.y; f[6] = b.z; f[7] = b.w;
  v8bf h = __builtin_convertvector(f, v8bf);
  return __builtin_bit_cast(uint4, h);
}
__device__ __forceinline__ float wred_sum(float v) {
  #pragma unroll
  for (int o = 16; o; o >>= 1) v += __shfl_xor(v, o);
  return v;
}
__device__ __forceinline__ float wred_max(float v) {
  #pragma unroll
  for (int o = 16; o; o >>= 1) v = fmaxf(v, __shfl_xor(v, o));
  return v;
}

// ---------------------------------------------------------------------------
// Problem constants
// ---------------------------------------------------------------------------
#define BATCH 64
#define NTOK  576
#define EDIM  1024
#define HIDD  512
#define NCLS  14
#define KASP  10
#define NQ    140         // NC*K
#define NQP   144         // padded to 16
#define NHEAD 8
#define DH    128
#define SCALE_QK 0.08838834764831845f   // 1/sqrt(128)

// ===========================================================================
// Generic GEMM: C = (A[MxK] @ W[NxK]^T + bias) * alpha   (row-major, K%32==0)
// fp32 in -> bf16 in LDS (double buffered) -> v_wmma_f32_16x16x32_bf16.
// Block 256 thr = 8 waves; tile 64x64; wave tile 16x32; 1 barrier / k-step.
//
// Ragged-M handling is branchless: out-of-range rows are CLAMPED to row M-1
// (valid duplicate data, no NaN risk). A garbage A-row only affects output
// row m>=M; a garbage W-row only affects output column n>=N; no n guard (N%64==0); rows are masked
// by the m<M store guard, so no zero-fill (and no exec branching) is needed.
// ===========================================================================
__global__ __launch_bounds__(256)
void gemm_xw_kernel(const float* __restrict__ A, const float* __restrict__ W,
                    const float* __restrict__ bias,
                    float* __restrict__ C, u16* __restrict__ Cb,
                    int M, int N, int K, float alpha, int relu)
{
  __shared__ __align__(16) u16 As[2][64][32];
  __shared__ __align__(16) u16 Bs[2][64][32];

  const int tid  = threadIdx.x;
  const int lane = tid & 31;
  const int wave = tid >> 5;
  const int m0 = blockIdx.y * 64;
  const int n0 = blockIdx.x * 64;
  const int wm = (wave & 3) * 16;
  const int wn = (wave >> 2) * 32;
  const int l15 = lane & 15;
  const int lhi = lane >> 4;

  const int lrow = tid >> 2;          // 0..63
  const int lcol = (tid & 3) * 8;     // 0,8,16,24

  const int gmr = m0 + lrow;
  const int gnr = n0 + lrow;
  const float* Aptr = A + (size_t)(gmr < M ? gmr : (M - 1)) * K + lcol;
  const float* Wptr = W + (size_t)(gnr < N ? gnr : (N - 1)) * K + lcol;

  v8f acc0 = {0.f,0.f,0.f,0.f,0.f,0.f,0.f,0.f};
  v8f acc1 = acc0;

  float4 fa0, fa1, fb0, fb1;

#define LOAD_TILES(k0) do {                                                   \
    const float4* pa_ = (const float4*)(Aptr + (k0));                         \
    const float4* pb_ = (const float4*)(Wptr + (k0));                         \
    fa0 = pa_[0]; fa1 = pa_[1];                                               \
    fb0 = pb_[0]; fb1 = pb_[1];                                               \
  } while (0)

#define STORE_TILES(buf) do {                                                 \
    *(uint4*)&As[buf][lrow][lcol] = cvt8(fa0, fa1);                           \
    *(uint4*)&Bs[buf][lrow][lcol] = cvt8(fb0, fb1);                           \
  } while (0)

#define COMPUTE(buf) do {                                                     \
    v16bf a_, b0_, b1_;                                                       \
    const uint4* ap_ = (const uint4*)&As[buf][wm + l15][0];                   \
    ((uint4*)&a_)[0] = ap_[lhi];                                              \
    ((uint4*)&a_)[1] = ap_[lhi + 2];                                          \
    const uint4* bp0_ = (const uint4*)&Bs[buf][wn + l15][0];                  \
    ((uint4*)&b0_)[0] = bp0_[lhi];                                            \
    ((uint4*)&b0_)[1] = bp0_[lhi + 2];                                        \
    const uint4* bp1_ = (const uint4*)&Bs[buf][wn + 16 + l15][0];             \
    ((uint4*)&b1_)[0] = bp1_[lhi];                                            \
    ((uint4*)&b1_)[1] = bp1_[lhi + 2];                                        \
    acc0 = WMMA_BF16(a_, b0_, acc0);                                          \
    acc1 = WMMA_BF16(a_, b1_, acc1);                                          \
  } while (0)

  // Prologue: fill buffer 0.
  LOAD_TILES(0);
  STORE_TILES(0);
  __syncthreads();

  int buf = 0;
  for (int k0 = 32; k0 < K; k0 += 32) {
    // Prefetch two k-steps ahead (global_prefetch_b8) to deepen the pipe.
    if (k0 + 32 < K) {
      __builtin_prefetch(Aptr + k0 + 32, 0, 3);
      __builtin_prefetch(Wptr + k0 + 32, 0, 3);
    }
    LOAD_TILES(k0);        // global loads for next tile (in flight)
    COMPUTE(buf);          // WMMA on current LDS buffer
    STORE_TILES(buf ^ 1);  // convert + stage next tile
    __syncthreads();
    buf ^= 1;
  }
  COMPUTE(buf);

#undef LOAD_TILES
#undef STORE_TILES
#undef COMPUTE

  // --- epilogue: C/D layout VGPR i -> M = i + lhi*8, N = l15 ---
  #pragma unroll
  for (int t = 0; t < 2; ++t) {
    const v8f acc = t ? acc1 : acc0;
    const int n = n0 + wn + t*16 + l15;
    const float bv = bias ? bias[n] : 0.f;
    #pragma unroll
    for (int i = 0; i < 8; ++i) {
      const int m = m0 + wm + lhi*8 + i;
      if (m < M) {
        float v = (acc[i] + bv) * alpha;
        if (relu) v = fmaxf(v, 0.f);
        if (C)  C [(size_t)m * N + n] = v;
        if (Cb) Cb[(size_t)m * N + n] = f2bf(v);
      }
    }
  }
}

// ===========================================================================
// scores: S[b,h,q,n] = sum_d qh[q, h*128+d] * kh[b,n, h*128+d]
// qh bf16 (pre-scaled by 1/sqrt(dh)), kv bf16 [b][n][2048] (k | v).
// Grid (9 ntiles of 64, 9 qtiles of 16, B*NH); block 128 = 4 waves.
// ===========================================================================
__global__ __launch_bounds__(128)
void scores_kernel(const u16* __restrict__ qh, const u16* __restrict__ kv,
                   float* __restrict__ S)
{
  const int lane = threadIdx.x & 31;
  const int wave = threadIdx.x >> 5;
  const int bh = blockIdx.z;
  const int b  = bh >> 3, h = bh & 7;
  const int q0 = blockIdx.y * 16;
  const int n0 = blockIdx.x * 64 + wave * 16;
  const int l15 = lane & 15, lhi = lane >> 4;

  v8f acc = {0.f,0.f,0.f,0.f,0.f,0.f,0.f,0.f};
  const u16* qrow = qh + (size_t)(q0 + l15) * EDIM + h * DH;
  const u16* krow = kv + ((size_t)b * NTOK + n0 + l15) * 2048 + h * DH;

  #pragma unroll
  for (int k0 = 0; k0 < DH; k0 += 32) {
    v16bf a, bf;
    const uint4* ap = (const uint4*)(qrow + k0);
    ((uint4*)&a)[0] = ap[lhi];
    ((uint4*)&a)[1] = ap[lhi + 2];
    const uint4* bp = (const uint4*)(krow + k0);
    ((uint4*)&bf)[0] = bp[lhi];
    ((uint4*)&bf)[1] = bp[lhi + 2];
    acc = WMMA_BF16(a, bf, acc);
  }
  const int nc = n0 + l15;
  #pragma unroll
  for (int i = 0; i < 8; ++i) {
    const int q = q0 + lhi*8 + i;
    S[((size_t)bh * NQP + q) * NTOK + nc] = acc[i];
  }
}

// ===========================================================================
// softmax over n (576) per (b,h,q<140); fp32 back into S, bf16 into P.
// 1 wave per row; block 256 = 8 rows.
// ===========================================================================
__global__ __launch_bounds__(256)
void softmax_kernel(float* __restrict__ S, u16* __restrict__ P)
{
  const int row = blockIdx.x * 8 + (threadIdx.x >> 5);
  const int lane = threadIdx.x & 31;
  if (row >= BATCH * NHEAD * NQ) return;
  const int q  = row % NQ;
  const int bh = row / NQ;
  float* s = S + ((size_t)bh * NQP + q) * NTOK;
  u16*   p = P + ((size_t)bh * NQP + q) * NTOK;

  float vals[18];
  float m = -3.4e38f;
  #pragma unroll
  for (int i = 0; i < 18; ++i) { vals[i] = s[lane + i*32]; m = fmaxf(m, vals[i]); }
  m = wred_max(m);
  float sum = 0.f;
  #pragma unroll
  for (int i = 0; i < 18; ++i) { vals[i] = __expf(vals[i] - m); sum += vals[i]; }
  sum = wred_sum(sum);
  const float inv = 1.f / sum;
  #pragma unroll
  for (int i = 0; i < 18; ++i) {
    const float v = vals[i] * inv;
    s[lane + i*32] = v;
    p[lane + i*32] = f2bf(v);
  }
}

// mean over heads -> attention_weights output (B, NC, K, N)
__global__ __launch_bounds__(256)
void mean_attn_kernel(const float* __restrict__ S, float* __restrict__ out)
{
  const size_t idx = (size_t)blockIdx.x * 256 + threadIdx.x;
  if (idx >= (size_t)BATCH * NQ * NTOK) return;
  const int n = (int)(idx % NTOK);
  const size_t t = idx / NTOK;
  const int q = (int)(t % NQ);
  const int b = (int)(t / NQ);
  float acc = 0.f;
  #pragma unroll
  for (int h = 0; h < NHEAD; ++h)
    acc += S[(((size_t)(b*NHEAD + h)) * NQP + q) * NTOK + n];
  out[idx] = acc * 0.125f;
}

// ===========================================================================
// P @ V : att[b, q, h*128+d] = sum_n P[b,h,q,n] * vh[b,n,h*128+d]
// Grid (2 dblocks of 64, 9 qtiles, B*NH); block 128 = 4 waves; LDS-transposed V.
// ===========================================================================
__global__ __launch_bounds__(128)
void pv_kernel(const u16* __restrict__ P, const u16* __restrict__ kv,
               float* __restrict__ att)
{
  __shared__ __align__(16) u16 Vt[64][32];   // [d-local][k-local]
  const int lane = threadIdx.x & 31;
  const int wave = threadIdx.x >> 5;
  const int bh = blockIdx.z, b = bh >> 3, h = bh & 7;
  const int q0 = blockIdx.y * 16;
  const int d0 = blockIdx.x * 64;
  const int l15 = lane & 15, lhi = lane >> 4;

  const int grow = threadIdx.x >> 2;          // k-local 0..31
  const int gseg = (threadIdx.x & 3) * 16;    // d-local segment

  v8f acc = {0.f,0.f,0.f,0.f,0.f,0.f,0.f,0.f};
  const u16* prow = P + ((size_t)bh * NQP + q0 + l15) * NTOK;

  for (int k0 = 0; k0 < NTOK; k0 += 32) {
    // coalesced 32B load of vh rows, then transpose into LDS
    const u16* src = kv + ((size_t)b * NTOK + k0 + grow) * 2048 + EDIM + h*DH + d0 + gseg;
    uint4 u0 = ((const uint4*)src)[0];
    uint4 u1 = ((const uint4*)src)[1];
    u16 tmp[16];
    *(uint4*)&tmp[0] = u0; *(uint4*)&tmp[8] = u1;
    __syncthreads();
    #pragma unroll
    for (int i = 0; i < 16; ++i) Vt[gseg + i][grow] = tmp[i];
    __syncthreads();

    v16bf a, bf;
    const uint4* ap = (const uint4*)(prow + k0);
    ((uint4*)&a)[0] = ap[lhi];
    ((uint4*)&a)[1] = ap[lhi + 2];
    const uint4* bp = (const uint4*)&Vt[wave*16 + l15][0];
    ((uint4*)&bf)[0] = bp[lhi];
    ((uint4*)&bf)[1] = bp[lhi + 2];
    acc = WMMA_BF16(a, bf, acc);
  }
  const int dcol = h*DH + d0 + wave*16 + l15;
  #pragma unroll
  for (int i = 0; i < 8; ++i) {
    const int q = q0 + lhi*8 + i;
    att[((size_t)b * NQP + q) * EDIM + dcol] = acc[i];
  }
}

// attended_mean[b*NC+j][e] = mean_k att[b][j*10+k][e]
__global__ __launch_bounds__(256)
void meanK_kernel(const float* __restrict__ att, float* __restrict__ out)
{
  const size_t idx = (size_t)blockIdx.x * 256 + threadIdx.x;
  if (idx >= (size_t)BATCH * NCLS * EDIM) return;
  const int e  = (int)(idx & (EDIM - 1));
  const int bj = (int)(idx >> 10);
  const int j = bj % NCLS, b = bj / NCLS;
  float acc = 0.f;
  #pragma unroll
  for (int k = 0; k < KASP; ++k)
    acc += att[((size_t)b * NQP + j*KASP + k) * EDIM + e];
  out[idx] = acc * 0.1f;
}

// LayerNorm rows of 1024 (1 wave / row)
__global__ __launch_bounds__(256)
void ln_kernel(const float* __restrict__ x, const float* __restrict__ g,
               const float* __restrict__ bta, float* __restrict__ out, int rows)
{
  const int row = blockIdx.x * 8 + (threadIdx.x >> 5);
  const int lane = threadIdx.x & 31;
  if (row >= rows) return;
  const float* xr = x + (size_t)row * EDIM;
  float s = 0.f;
  #pragma unroll
  for (int i = 0; i < 32; ++i) s += xr[lane + i*32];
  const float mean = wred_sum(s) * (1.f / EDIM);
  float v = 0.f;
  #pragma unroll
  for (int i = 0; i < 32; ++i) { const float d = xr[lane + i*32] - mean; v += d*d; }
  const float rstd = rsqrtf(wred_sum(v) * (1.f / EDIM) + 1e-5f);
  #pragma unroll
  for (int i = 0; i < 32; ++i) {
    const int e = lane + i*32;
    out[(size_t)row * EDIM + e] = (xr[e] - mean) * rstd * g[e] + bta[e];
  }
}

// contrastive similarity per batch row (1 wave / row)
__global__ __launch_bounds__(256)
void sim_kernel(const float* __restrict__ cout, const float* __restrict__ tp,
                float* __restrict__ sim)
{
  const int row = blockIdx.x * 8 + (threadIdx.x >> 5);
  const int lane = threadIdx.x & 31;
  if (row >= BATCH) return;
  const float* c = cout + (size_t)row * EDIM;
  const float* t = tp   + (size_t)row * EDIM;
  float cc = 0.f, tt = 0.f, ct = 0.f;
  #pragma unroll
  for (int i = 0; i < 32; ++i) {
    const float cv = c[lane + i*32], tv = t[lane + i*32];
    cc += cv*cv; tt += tv*tv; ct += cv*tv;
  }
  cc = wred_sum(cc); tt = wred_sum(tt); ct = wred_sum(ct);
  if (lane == 0) {
    const float nc = fmaxf(sqrtf(cc), 1e-12f);
    const float nt = fmaxf(sqrtf(tt), 1e-12f);
    sim[row] = ct / (nc * nt) * (1.f / 0.1f);
  }
}

// final logits: LN+per-class dot (cross), direct-MLP second layer, combine.
__global__ __launch_bounds__(256)
void final_kernel(const float* __restrict__ ap,      // attended_proj [896][1024]
                  const float* __restrict__ cg, const float* __restrict__ cb,
                  const float* __restrict__ fw, const float* __restrict__ fb,
                  const float* __restrict__ h1,      // [64][512]
                  const float* __restrict__ w2, const float* __restrict__ b2,
                  const float* __restrict__ sim,
                  float* __restrict__ out)           // d_out base
{
  const int w = blockIdx.x * 8 + (threadIdx.x >> 5);
  const int lane = threadIdx.x & 31;
  if (w >= BATCH * NCLS) return;
  const int j = w % NCLS, b = w / NCLS;
  const float* xr = ap + (size_t)w * EDIM;

  float s = 0.f;
  #pragma unroll
  for (int i = 0; i < 32; ++i) s += xr[lane + i*32];
  const float mean = wred_sum(s) * (1.f / EDIM);
  float v = 0.f;
  #pragma unroll
  for (int i = 0; i < 32; ++i) { const float d = xr[lane + i*32] - mean; v += d*d; }
  const float rstd = rsqrtf(wred_sum(v) * (1.f / EDIM) + 1e-5f);

  float lc = 0.f;
  #pragma unroll
  for (int i = 0; i < 32; ++i) {
    const int e = lane + i*32;
    lc += ((xr[e] - mean) * rstd * cg[e] + cb[e]) * fw[(size_t)j * EDIM + e];
  }
  lc = wred_sum(lc);

  float ld = 0.f;
  #pragma unroll
  for (int i = 0; i < 16; ++i) {
    const int e = lane + i*32;
    ld += h1[(size_t)b * HIDD + e] * w2[(size_t)j * HIDD + e];
  }
  ld = wred_sum(ld);

  if (lane == 0) {
    const float logits_cross  = lc + fb[j];
    const float logits_direct = ld + b2[j];
    const float sup  = 0.5f * (logits_cross + logits_direct);
    const float cont = sim[b];
    out[w]                 = cont;                 // contrastive_logits
    out[BATCH*NCLS + w]    = sup;                  // supervised_logits
    out[2*BATCH*NCLS + w]  = 0.5f * (sup + cont);  // final_logits
  }
}

// ===========================================================================
// Host launcher
// ===========================================================================
extern "C" void kernel_launch(void* const* d_in, const int* in_sizes, int n_in,
                              void* d_out, int out_size, void* d_ws, size_t ws_size,
                              hipStream_t stream) {
  const float* vision_features = (const float*)d_in[0];
  const float* text_embeddings = (const float*)d_in[1];
  const float* vision_global   = (const float*)d_in[2];
  const float* aspect_queries  = (const float*)d_in[3];
  const float* ch_in_w  = (const float*)d_in[4];
  const float* ch_in_b  = (const float*)d_in[5];
  const float* ch_out_w = (const float*)d_in[6];
  const float* ch_out_b = (const float*)d_in[7];
  const float* tp_w     = (const float*)d_in[8];
  const float* tp_b     = (const float*)d_in[9];
  const float* ca_in_w  = (const float*)d_in[10];
  const float* ca_in_b  = (const float*)d_in[11];
  const float* ca_out_w = (const float*)d_in[12];
  const float* ca_out_b = (const float*)d_in[13];
  const float* cross_ln_g = (const float*)d_in[14];
  const float* cross_ln_b = (const float*)d_in[15];
  const float* cross_fc_w = (const float*)d_in[16];
  const float* cross_fc_b = (const float*)d_in[17];
  const float* dir_ln_g = (const float*)d_in[18];
  const float* dir_ln_b = (const float*)d_in[19];
  const float* dir_w1   = (const float*)d_in[20];
  const float* dir_b1   = (const float*)d_in[21];
  const float* dir_w2   = (const float*)d_in[22];
  const float* dir_b2   = (const float*)d_in[23];
  float* out = (float*)d_out;

  // ---- workspace carve-up ----
  char* base = (char*)d_ws;
  size_t o = 0;
  auto carve = [&](size_t bytes) { char* p = base + o; o = (o + bytes + 255) & ~(size_t)255; return p; };
  float* tp_f   = (float*)carve((size_t)BATCH*EDIM*4);
  float* cv_f   = (float*)carve((size_t)BATCH*EDIM*4);
  float* cout_f = (float*)carve((size_t)BATCH*EDIM*4);
  float* g_f    = (float*)carve((size_t)BATCH*EDIM*4);
  float* h1_f   = (float*)carve((size_t)BATCH*HIDD*4);
  float* sim_f  = (float*)carve((size_t)BATCH*4);
  float* am_f   = (float*)carve((size_t)BATCH*NCLS*EDIM*4);
  float* apj_f  = (float*)carve((size_t)BATCH*NCLS*EDIM*4);
  u16*   qh_b   = (u16*)  carve((size_t)NQP*EDIM*2);
  u16*   kv_b   = (u16*)  carve((size_t)BATCH*NTOK*2048*2);
  float* S_f    = (float*)carve((size_t)BATCH*NHEAD*NQP*NTOK*4);
  u16*   P_b    = (u16*)  carve((size_t)BATCH*NHEAD*NQP*NTOK*2);
  float* att_f  = (float*)carve((size_t)BATCH*NQP*EDIM*4);
  (void)ws_size; (void)n_in; (void)in_sizes; (void)out_size;

  // 1) LN(vision_global) -> g
  ln_kernel<<<dim3((BATCH+7)/8), 256, 0, stream>>>(vision_global, dir_ln_g, dir_ln_b, g_f, BATCH);
  // 2) text_proj = text @ tp_w^T + tp_b
  gemm_xw_kernel<<<dim3(EDIM/64, BATCH/64), 256, 0, stream>>>(
      text_embeddings, tp_w, tp_b, tp_f, nullptr, BATCH, EDIM, EDIM, 1.f, 0);
  // 3) contrastive v-proj (softmax over 1 key == 1, q/k projections dead)
  gemm_xw_kernel<<<dim3(EDIM/64, BATCH/64), 256, 0, stream>>>(
      tp_f, ch_in_w + (size_t)2*EDIM*EDIM, ch_in_b + 2*EDIM, cv_f, nullptr, BATCH, EDIM, EDIM, 1.f, 0);
  // 4) contrastive out-proj
  gemm_xw_kernel<<<dim3(EDIM/64, BATCH/64), 256, 0, stream>>>(
      cv_f, ch_out_w, ch_out_b, cout_f, nullptr, BATCH, EDIM, EDIM, 1.f, 0);
  // 5) qh = (aspect_queries @ wq^T + bq) * 1/sqrt(dh)  -> bf16
  gemm_xw_kernel<<<dim3(EDIM/64, (NQ+63)/64), 256, 0, stream>>>(
      aspect_queries, ca_in_w, ca_in_b, nullptr, qh_b, NQ, EDIM, EDIM, SCALE_QK, 0);
  // 6) K/V projection (the big GEMM): kv = features @ [wk;wv]^T + [bk;bv] -> bf16
  gemm_xw_kernel<<<dim3(2048/64, (BATCH*NTOK)/64), 256, 0, stream>>>(
      vision_features, ca_in_w + (size_t)EDIM*EDIM, ca_in_b + EDIM,
      nullptr, kv_b, BATCH*NTOK, 2048, EDIM, 1.f, 0);
  // 7) direct MLP layer 1 (relu)
  gemm_xw_kernel<<<dim3(HIDD/64, BATCH/64), 256, 0, stream>>>(
      g_f, dir_w1, dir_b1, h1_f, nullptr, BATCH, HIDD, EDIM, 1.f, 1);
  // 8) scores
  scores_kernel<<<dim3(NTOK/64, NQP/16, BATCH*NHEAD), 128, 0, stream>>>(qh_b, kv_b, S_f);
  // 9) softmax (fp32 back + bf16 P)
  softmax_kernel<<<dim3((BATCH*NHEAD*NQ + 7)/8), 256, 0, stream>>>(S_f, P_b);
  // 10) attention_weights output = mean over heads
  mean_attn_kernel<<<dim3((unsigned)(((size_t)BATCH*NQ*NTOK + 255)/256)), 256, 0, stream>>>(
      S_f, out + 3*BATCH*NCLS);
  // 11) att = P @ V
  pv_kernel<<<dim3(DH/64, NQP/16, BATCH*NHEAD), 128, 0, stream>>>(P_b, kv_b, att_f);
  // 12) mean over K aspects (before out-proj: linear fold)
  meanK_kernel<<<dim3((unsigned)(((size_t)BATCH*NCLS*EDIM + 255)/256)), 256, 0, stream>>>(att_f, am_f);
  // 13) attended out-proj
  gemm_xw_kernel<<<dim3(EDIM/64, (BATCH*NCLS)/64), 256, 0, stream>>>(
      am_f, ca_out_w, ca_out_b, apj_f, nullptr, BATCH*NCLS, EDIM, EDIM, 1.f, 0);
  // 14) contrastive similarity
  sim_kernel<<<dim3((BATCH+7)/8), 256, 0, stream>>>(cout_f, tp_f, sim_f);
  // 15) finalize all three logit tensors
  final_kernel<<<dim3((BATCH*NCLS+7)/8), 256, 0, stream>>>(
      apj_f, cross_ln_g, cross_ln_b, cross_fc_w, cross_fc_b,
      h1_f, dir_w2, dir_b2, sim_f, out);
}
